// _NN_for_LC_INC_88252987998248
// MI455X (gfx1250) — matrix operand order repaired and checked by reading, not compiled, not run on hardware
//
#include <hip/hip_runtime.h>
#include <hip/hip_bf16.h>
#include <math.h>
#include <stdint.h>

typedef __attribute__((ext_vector_type(16))) __bf16 v16bf;
typedef __attribute__((ext_vector_type(8)))  __bf16 v8bf;
typedef __attribute__((ext_vector_type(8)))  float  v8f;
typedef __attribute__((ext_vector_type(4)))  int    v4i;

#define TILE_M 128
#define TILE_N 128
#define TILE_K 32
#define PAD    40   // halfs per LDS row: 80B stride, 16B-aligned chunks, conflict-light

#define ASG __attribute__((address_space(1)))
#define ASL __attribute__((address_space(3)))

#if __has_builtin(__builtin_amdgcn_global_load_async_to_lds_b128)
#define HAS_ASYNC_LDS 1
#else
#define HAS_ASYNC_LDS 0
#endif

__device__ __forceinline__ void wait_async_zero() {
#if HAS_ASYNC_LDS
#if __has_builtin(__builtin_amdgcn_s_wait_asynccnt)
    __builtin_amdgcn_s_wait_asynccnt(0);
#else
    asm volatile("s_wait_asynccnt 0" ::: "memory");
#endif
#endif
}

union FragBF {
    v16bf v;
    v8bf  h[2];
};

// ---------------------------------------------------------------------------
// C[M,N](bf16) = LeakyReLU(A[M,K](bf16) @ Wt[N,K](bf16)^T + bias(f32), slope)
// Double-buffered LDS tiles filled by global_load_async_to_lds_b128 (no VGPR
// staging, ASYNCcnt-tracked); v_wmma_f32_16x16x32_bf16 with f32 accumulation.
// ---------------------------------------------------------------------------
__global__ __launch_bounds__(256)
void gemm_bf16_bias_lrelu(const __bf16* __restrict__ A,
                          const __bf16* __restrict__ Wt,   // [N,K] pre-transposed
                          const float*  __restrict__ bias,
                          __bf16* __restrict__ C,
                          int M, int N, int K, float slope)
{
    __shared__ __bf16 As[2][TILE_M * PAD];
    __shared__ __bf16 Ws[2][TILE_N * PAD];

    const int tid  = threadIdx.x;
    const int lane = tid & 31;
    const int wave = tid >> 5;
    const int m0   = blockIdx.y * TILE_M;
    const int n0   = blockIdx.x * TILE_N;

    v8f acc[8] = {};

    // 128 rows x 32 halfs = 512 16B-chunks per tile side; 2 chunks/thread/side.
    auto stage = [&](int k0, int buf) {
#pragma unroll
        for (int i = 0; i < 2; ++i) {
            int chunk = i * 256 + tid;
            int row = chunk >> 2, c8 = (chunk & 3) * 8;
            const __bf16* ga = A  + (size_t)(m0 + row) * K + k0 + c8;
            const __bf16* gw = Wt + (size_t)(n0 + row) * K + k0 + c8;
            __bf16* la = &As[buf][row * PAD + c8];
            __bf16* lw = &Ws[buf][row * PAD + c8];
#if HAS_ASYNC_LDS
            // Direct global->LDS DMA, no VGPR data path, tracked by ASYNCcnt.
            __builtin_amdgcn_global_load_async_to_lds_b128(
                (ASG v4i*)(uintptr_t)ga, (ASL v4i*)(unsigned)(uintptr_t)la, 0, 0);
            __builtin_amdgcn_global_load_async_to_lds_b128(
                (ASG v4i*)(uintptr_t)gw, (ASL v4i*)(unsigned)(uintptr_t)lw, 0, 0);
#else
            // Fallback: transient vector copy (no liveness across the WMMA loop).
            uint4 va = *(const uint4*)ga;
            uint4 vw = *(const uint4*)gw;
            *(uint4*)la = va;
            *(uint4*)lw = vw;
#endif
        }
    };

    auto compute = [&](int buf) {
        // A frag (16-bit A 16x32 ISA layout): lane L = row M=L&15,
        // chunks [c*8, c*8+8) and [c*8+16, c*8+24), c = L>>4.
        FragBF a;
        const __bf16* ap = &As[buf][(wave * 16 + (lane & 15)) * PAD + ((lane >> 4) * 8)];
        a.h[0] = *(const v8bf*)(ap);
        a.h[1] = *(const v8bf*)(ap + 16);
#pragma unroll
        for (int nt = 0; nt < 8; ++nt) {
            // B frag (16-bit B 32x16): lane L = col N=L&15, K chunks [c*8, c*8+8) and [c*8+16, c*8+24), c = L>>4.
            FragBF b;
            const __bf16* bp = &Ws[buf][(nt * 16 + (lane & 15)) * PAD + ((lane >> 4) * 8)];
            b.h[0] = *(const v8bf*)(bp);
            b.h[1] = *(const v8bf*)(bp + 16);
            acc[nt] = __builtin_amdgcn_wmma_f32_16x16x32_bf16(
                false, a.v, false, b.v, (short)0, acc[nt], false, false);
        }
    };

    const int nk = K / TILE_K;

    stage(0, 0);
    wait_async_zero();
    __syncthreads();

    for (int kt = 0; kt < nk; ++kt) {
        const int cur = kt & 1;
        if (kt + 1 < nk) {
            stage((kt + 1) * TILE_K, cur ^ 1);   // fills overlap the WMMAs below
            if (kt + 2 < nk) {
                __builtin_prefetch(A  + (size_t)(m0 + (tid >> 1)) * K + (kt + 2) * TILE_K, 0, 0);
                __builtin_prefetch(Wt + (size_t)(n0 + (tid >> 1)) * K + (kt + 2) * TILE_K, 0, 0);
            }
        }
        compute(cur);
        if (kt + 1 < nk) {
            wait_async_zero();     // this wave's fills of buf cur^1 have landed
            __syncthreads();       // everyone's fills visible
        }
    }

    // Epilogue. D layout: VGPR r, lane L -> M = r + 8*(L>>4), N = L&15.
    const int mrow = m0 + wave * 16 + ((lane >> 4) << 3);
#pragma unroll
    for (int nt = 0; nt < 8; ++nt) {
        int n = n0 + nt * 16 + (lane & 15);
        float bn = bias[n];
#pragma unroll
        for (int r = 0; r < 8; ++r) {
            float v = acc[nt][r] + bn;
            v = (v >= 0.f) ? v : v * slope;
            C[(size_t)(mrow + r) * N + n] = (__bf16)v;
        }
    }
}

// ---------------------------------------------------------------------------
// Elementwise f32 -> bf16 (vectorized: float4 in, 4x bf16 out)
// ---------------------------------------------------------------------------
__global__ __launch_bounds__(256)
void cvt_f32_bf16(const float* __restrict__ in, __bf16* __restrict__ out, size_t n4)
{
    size_t i = (size_t)blockIdx.x * 256 + threadIdx.x;
    if (i >= n4) return;
    float4 v = ((const float4*)in)[i];
    union { __bf16 h[4]; unsigned long long u; } p;
    p.h[0] = (__bf16)v.x; p.h[1] = (__bf16)v.y;
    p.h[2] = (__bf16)v.z; p.h[3] = (__bf16)v.w;
    ((unsigned long long*)out)[i] = p.u;
}

// ---------------------------------------------------------------------------
// W[K,N] f32 -> Wt[N,K] bf16 (32x32 LDS tiles, conflict-free via pad)
// ---------------------------------------------------------------------------
__global__ __launch_bounds__(256)
void transpose_cvt_bf16(const float* __restrict__ W, __bf16* __restrict__ Wt,
                        int Kdim, int Ndim)
{
    __shared__ float tile[32][33];
    const int k0 = blockIdx.y * 32, n0 = blockIdx.x * 32;
    const int tx = threadIdx.x & 31, ty = threadIdx.x >> 5;   // ty in 0..7
#pragma unroll
    for (int i = ty; i < 32; i += 8)
        tile[i][tx] = W[(size_t)(k0 + i) * Ndim + (n0 + tx)];
    __syncthreads();
#pragma unroll
    for (int i = ty; i < 32; i += 8)
        Wt[(size_t)(n0 + i) * Kdim + (k0 + tx)] = (__bf16)tile[tx][i];
}

// ---------------------------------------------------------------------------
// Compact bilinear pooling: count-sketch has 1 nonzero/row, so the FFT
// circular convolution collapses to a 128x128 outer-product scatter:
//   cbp[(h1[k]+h2[l]) & 2047] += s1[k]*ex[k] * s2[l]*ec[l]   (ds_add_f32)
// ---------------------------------------------------------------------------
__global__ __launch_bounds__(256)
void cbp_sketch_conv(const __bf16* __restrict__ ex, const __bf16* __restrict__ ec,
                     const int* __restrict__ h1, const float* __restrict__ s1,
                     const int* __restrict__ h2, const float* __restrict__ s2,
                     __bf16* __restrict__ cbp)
{
    __shared__ float bins[2048];
    __shared__ float av[128], cv[128];
    __shared__ int   ha[128], hb[128];

    const int b = blockIdx.x;
    const int t = threadIdx.x;

    for (int i = t; i < 2048; i += 256) bins[i] = 0.f;
    if (t < 128) {
        av[t] = s1[t] * (float)ex[(size_t)b * 128 + t];
        ha[t] = h1[t];
    } else {
        int l = t - 128;
        cv[l] = s2[l] * (float)ec[(size_t)b * 128 + l];
        hb[l] = h2[l];
    }
    __syncthreads();

    for (int p = t; p < 128 * 128; p += 256) {
        int k = p >> 7, l = p & 127;
        atomicAdd(&bins[(ha[k] + hb[l]) & 2047], av[k] * cv[l]);
    }
    __syncthreads();

    for (int i = t; i < 2048; i += 256)
        cbp[(size_t)b * 2048 + i] = (__bf16)bins[i];
}

// ---------------------------------------------------------------------------
// logits = Z[b,1024](bf16) @ Wc2[1024,10] + bc2; softmax over 10 classes (f32 out)
// ---------------------------------------------------------------------------
__global__ __launch_bounds__(256)
void head_softmax(const __bf16* __restrict__ Z, const float* __restrict__ Wc2,
                  const float* __restrict__ bc2, float* __restrict__ out)
{
    __shared__ float part[10];
    const int b = blockIdx.x;
    const int t = threadIdx.x;

    if (t < 10) part[t] = bc2[t];
    __syncthreads();

    float acc[10];
#pragma unroll
    for (int c = 0; c < 10; ++c) acc[c] = 0.f;

    for (int i = t; i < 1024; i += 256) {
        float z = (float)Z[(size_t)b * 1024 + i];
#pragma unroll
        for (int c = 0; c < 10; ++c) acc[c] += z * Wc2[i * 10 + c];
    }
#pragma unroll
    for (int c = 0; c < 10; ++c) atomicAdd(&part[c], acc[c]);
    __syncthreads();

    if (t == 0) {
        float mx = part[0];
#pragma unroll
        for (int c = 1; c < 10; ++c) mx = fmaxf(mx, part[c]);
        float e[10], s = 0.f;
#pragma unroll
        for (int c = 0; c < 10; ++c) { e[c] = __expf(part[c] - mx); s += e[c]; }
        float inv = 1.f / s;
#pragma unroll
        for (int c = 0; c < 10; ++c) out[(size_t)b * 10 + c] = e[c] * inv;
    }
}

extern "C" void kernel_launch(void* const* d_in, const int* in_sizes, int n_in,
                              void* d_out, int out_size, void* d_ws, size_t ws_size,
                              hipStream_t stream) {
    const float* X   = (const float*)d_in[0];
    const float* Cen = (const float*)d_in[1];
    const float* W1  = (const float*)d_in[2];
    const float* b1  = (const float*)d_in[3];
    const float* W2  = (const float*)d_in[4];
    const float* b2  = (const float*)d_in[5];
    const float* W3  = (const float*)d_in[6];
    const float* b3  = (const float*)d_in[7];
    const int*   h1  = (const int*)d_in[8];
    const float* s1  = (const float*)d_in[9];
    const int*   h2  = (const int*)d_in[10];
    const float* s2  = (const float*)d_in[11];
    const float* Wc1 = (const float*)d_in[12];
    const float* bc1 = (const float*)d_in[13];
    const float* Wc2 = (const float*)d_in[14];
    const float* bc2 = (const float*)d_in[15];
    float* out = (float*)d_out;

    const int B = 16384;
    const float SLOPE = 0.2f;

    // Workspace layout, all bf16 (element counts)
    __bf16* ws   = (__bf16*)d_ws;
    __bf16* Xb   = ws;                            // B*1024
    __bf16* Cb   = Xb   + (size_t)B * 1024;       // B*1024
    __bf16* W1t  = Cb   + (size_t)B * 1024;       // 512*1024
    __bf16* W2t  = W1t  + (size_t)512 * 1024;     // 256*512
    __bf16* W3t  = W2t  + (size_t)256 * 512;      // 128*256
    __bf16* Wc1t = W3t  + (size_t)128 * 256;      // 1024*2048
    __bf16* t1   = Wc1t + (size_t)1024 * 2048;    // B*512
    __bf16* t2   = t1   + (size_t)B * 512;        // B*256
    __bf16* ex   = t2   + (size_t)B * 256;        // B*128
    __bf16* ec   = ex   + (size_t)B * 128;        // B*128
    __bf16* cbp  = ec   + (size_t)B * 128;        // B*2048
    __bf16* z    = cbp  + (size_t)B * 2048;       // B*1024

    dim3 blk(256);
    auto grid  = [](int M, int N) { return dim3(N / TILE_N, M / TILE_M); };
    auto tgrid = [](int Kd, int Nd) { return dim3(Nd / 32, Kd / 32); };

    // ---- one-time precision/layout prep (amortized across 128 M-tiles) ----
    cvt_f32_bf16<<<(B * 1024 / 4 + 255) / 256, blk, 0, stream>>>(X,   Xb, (size_t)B * 1024 / 4);
    cvt_f32_bf16<<<(B * 1024 / 4 + 255) / 256, blk, 0, stream>>>(Cen, Cb, (size_t)B * 1024 / 4);
    transpose_cvt_bf16<<<tgrid(1024, 512),  blk, 0, stream>>>(W1,  W1t,  1024, 512);
    transpose_cvt_bf16<<<tgrid(512,  256),  blk, 0, stream>>>(W2,  W2t,  512,  256);
    transpose_cvt_bf16<<<tgrid(256,  128),  blk, 0, stream>>>(W3,  W3t,  256,  128);
    transpose_cvt_bf16<<<tgrid(2048, 1024), blk, 0, stream>>>(Wc1, Wc1t, 2048, 1024);

    // ---- embed X: 1024 -> 512 -> 256 -> 128 ----
    gemm_bf16_bias_lrelu<<<grid(B, 512),  blk, 0, stream>>>(Xb, W1t, b1, t1, B, 512,  1024, SLOPE);
    gemm_bf16_bias_lrelu<<<grid(B, 256),  blk, 0, stream>>>(t1, W2t, b2, t2, B, 256,  512,  SLOPE);
    gemm_bf16_bias_lrelu<<<grid(B, 128),  blk, 0, stream>>>(t2, W3t, b3, ex, B, 128,  256,  SLOPE);
    // ---- embed Centers ----
    gemm_bf16_bias_lrelu<<<grid(B, 512),  blk, 0, stream>>>(Cb, W1t, b1, t1, B, 512,  1024, SLOPE);
    gemm_bf16_bias_lrelu<<<grid(B, 256),  blk, 0, stream>>>(t1, W2t, b2, t2, B, 256,  512,  SLOPE);
    gemm_bf16_bias_lrelu<<<grid(B, 128),  blk, 0, stream>>>(t2, W3t, b3, ec, B, 128,  256,  SLOPE);

    // ---- compact bilinear pooling (sparse circular conv, no FFT) ----
    cbp_sketch_conv<<<B, blk, 0, stream>>>(ex, ec, h1, s1, h2, s2, cbp);

    // ---- comparator hidden layer: [B,2048] @ [2048,1024] ----
    gemm_bf16_bias_lrelu<<<grid(B, 1024), blk, 0, stream>>>(cbp, Wc1t, bc1, z, B, 1024, 2048, SLOPE);

    // ---- head + softmax ----
    head_softmax<<<B, blk, 0, stream>>>(z, Wc2, bc2, out);
}
